// Block_14577119003070
// MI455X (gfx1250) — matrix operand order repaired and checked by reading, not compiled, not run on hardware
//
#include <hip/hip_runtime.h>
#include <hip/hip_bf16.h>

// ---------------------------------------------------------------------------
// Types for CDNA5 WMMA (wave32): v16bf A/B fragments, v8f accumulator.
// ---------------------------------------------------------------------------
typedef __attribute__((ext_vector_type(16))) __bf16 bf16x16;
typedef __attribute__((ext_vector_type(8)))  float  f32x8;
typedef __attribute__((ext_vector_type(4)))  unsigned int u32x4;
typedef __attribute__((ext_vector_type(8)))  int    i32x8;
typedef __attribute__((ext_vector_type(4)))  int    i32x4;

union Frag {
    bf16x16 v;
    uint4   q[2];
};

#define WMMA_BF16(a, b, c) \
    __builtin_amdgcn_wmma_f32_16x16x32_bf16(false, (a), false, (b), (short)0, (c), false, false)

__device__ __forceinline__ unsigned short f2b(float f) {
    unsigned int u = __builtin_bit_cast(unsigned int, f);
    unsigned int r = (u + 0x7FFFu + ((u >> 16) & 1u)) >> 16;   // round-to-nearest-even
    return (unsigned short)r;
}

// Low 32 bits of a generic pointer to LDS == wave-relative LDS byte offset
// (addrspacecast(3->0) is {aperture_hi, lds_offset}).
__device__ __forceinline__ unsigned lds_off(const void* p) {
    return (unsigned)(size_t)p;
}

// ---------------------------------------------------------------------------
// TDM: issue a 2-D bf16 tile load (tile_k x tile_rows, row pitch tensor_k
// elements) into LDS with 16B padding every 64B -> LDS row pitch 40 elements.
// D# packing per CDNA5 ISA sec 8.3/8.4.  6-arg builtin (clang-23 lane).
// ---------------------------------------------------------------------------
__device__ __forceinline__ void tdm_load_tile(unsigned ldsaddr, const unsigned short* gptr,
                                              int tensor_k, int tensor_rows) {
    unsigned long long ga = (unsigned long long)(size_t)gptr;
    u32x4 g0;
    g0[0] = 1u;                                             // count=1, no gather
    g0[1] = ldsaddr;                                        // lds_addr
    g0[2] = (unsigned)(ga & 0xFFFFFFFFull);                 // global_addr[31:0]
    g0[3] = (unsigned)((ga >> 32) & 0x1FFFFFFull)           // global_addr[56:32]
            | (2u << 30);                                   // type=2 (image)
    i32x8 g1;
    g1[0] = (int)((1u << 16)                                // data_size = 2B
                  | (1u << 20)                              // pad_enable
                  | (3u << 22)                              // pad_interval: 16 DW = 64B
                  | (3u << 25));                            // pad_amount: 4 DW = 16B
    g1[1] = (int)(((unsigned)tensor_k & 0xFFFFu) << 16);    // tensor_dim0 lo16
    g1[2] = (int)(((unsigned)tensor_k >> 16)                // tensor_dim0 hi16
                  | (((unsigned)tensor_rows & 0xFFFFu) << 16)); // tensor_dim1 lo16
    g1[3] = (int)(((unsigned)tensor_rows >> 16)             // tensor_dim1 hi16
                  | (32u << 16));                           // tile_dim0 = 32 (k)
    g1[4] = 128;                                            // tile_dim1 = 128 rows, tile_dim2=0
    g1[5] = tensor_k;                                       // tensor_dim0_stride lo32
    g1[6] = 0;                                              // stride hi16, dim1_stride lo16
    g1[7] = 0;
    i32x4 z4 = {0, 0, 0, 0};
    i32x8 z8 = {0, 0, 0, 0, 0, 0, 0, 0};
    __builtin_amdgcn_tensor_load_to_lds(g0, g1, z4, z4, z8, 0);
}

static constexpr int B_ = 4;
static constexpr int T_ = 2048;
static constexpr int C_ = 1024;
static constexpr int H_ = 16;
static constexpr int D_ = 64;
static constexpr int ROWS = B_ * T_;          // 8192

// ---------------------------------------------------------------------------
// fp32 [K][N] -> bf16 transposed [N][K] (weights, once per launch)
// ---------------------------------------------------------------------------
__global__ void cvt_t_kernel(const float* __restrict__ in, unsigned short* __restrict__ out,
                             int Kd, int Nd) {
    int i = blockIdx.x * blockDim.x + threadIdx.x;
    if (i < Kd * Nd) {
        int k = i / Nd, n = i % Nd;
        out[(size_t)n * Kd + k] = f2b(in[i]);
    }
}

// ---------------------------------------------------------------------------
// LayerNorm over C=1024, one row per 256-thread block, bf16 output
// ---------------------------------------------------------------------------
__global__ __launch_bounds__(256) void ln_kernel(const float* __restrict__ x,
                                                 const float* __restrict__ g,
                                                 const float* __restrict__ bta,
                                                 unsigned short* __restrict__ out) {
    __shared__ float red[256];
    const int row = blockIdx.x;
    const int t   = threadIdx.x;
    const float* xr = x + (size_t)row * C_;
    float4 v = *(const float4*)(xr + t * 4);
    red[t] = v.x + v.y + v.z + v.w;
    __syncthreads();
    for (int off = 128; off > 0; off >>= 1) {
        if (t < off) red[t] += red[t + off];
        __syncthreads();
    }
    float mu = red[0] * (1.0f / C_);
    __syncthreads();
    float dx = v.x - mu, dy = v.y - mu, dz = v.z - mu, dw = v.w - mu;
    red[t] = dx * dx + dy * dy + dz * dz + dw * dw;
    __syncthreads();
    for (int off = 128; off > 0; off >>= 1) {
        if (t < off) red[t] += red[t + off];
        __syncthreads();
    }
    float rs = rsqrtf(red[0] * (1.0f / C_) + 1e-5f);
    float4 gg = *(const float4*)(g + t * 4);
    float4 bb = *(const float4*)(bta + t * 4);
    unsigned short* orow = out + (size_t)row * C_ + t * 4;
    orow[0] = f2b(dx * rs * gg.x + bb.x);
    orow[1] = f2b(dy * rs * gg.y + bb.y);
    orow[2] = f2b(dz * rs * gg.z + bb.z);
    orow[3] = f2b(dw * rs * gg.w + bb.w);
}

// ---------------------------------------------------------------------------
// BF16 WMMA GEMM with TDM-staged, double-buffered LDS tiles.
// out[M,N] = A[M,K] * Wt[N,K]^T + bias, fused epilogues.
// Block tile 128x128 (8 waves, each 32x64 = 2x4 grid of 16x16 WMMA tiles).
// MODE 0: bf16 out, +bias
// MODE 1: f32 out,  +bias +residual
// MODE 2: bf16 out, +bias, tanh-GELU
// ---------------------------------------------------------------------------
template <int MODE>
__global__ __launch_bounds__(256) void gemm_bf16(const unsigned short* __restrict__ A,
                                                 const unsigned short* __restrict__ Wt,
                                                 const float* __restrict__ bias,
                                                 const float* __restrict__ res,
                                                 float* __restrict__ outF,
                                                 unsigned short* __restrict__ outB,
                                                 int M, int N, int K) {
    __shared__ alignas(16) unsigned short As[2][128 * 40];   // [row][k], pitch 40 (TDM pad)
    __shared__ alignas(16) unsigned short Bs[2][128 * 40];   // [n][k],   pitch 40 (TDM pad)
    const int tid  = threadIdx.x;
    const int lane = tid & 31;
    const int wid  = tid >> 5;
    const int wr   = wid & 3;        // 4 wave rows * 32
    const int wc   = wid >> 2;       // 2 wave cols * 64
    const int l16  = lane & 15;
    const int lh   = lane >> 4;
    const int m0   = blockIdx.y * 128;
    const int n0   = blockIdx.x * 128;
    const bool issuer = (wid == 0);

    f32x8 acc[2][4];
#pragma unroll
    for (int tr = 0; tr < 2; ++tr)
#pragma unroll
        for (int tc = 0; tc < 4; ++tc)
#pragma unroll
            for (int i = 0; i < 8; ++i) acc[tr][tc][i] = 0.0f;

    const int nk = K / 32;
    if (issuer) {   // prologue: stage 0
        tdm_load_tile(lds_off(&As[0][0]), A + (size_t)m0 * K, K, 128);
        tdm_load_tile(lds_off(&Bs[0][0]), Wt + (size_t)n0 * K, K, 128);
    }

    for (int i = 0; i < nk; ++i) {
        const int cur = i & 1;
        if (issuer) {
            if (i + 1 < nk) {
                const int k1 = (i + 1) * 32;
                tdm_load_tile(lds_off(&As[cur ^ 1][0]), A + (size_t)m0 * K + k1, K, 128);
                tdm_load_tile(lds_off(&Bs[cur ^ 1][0]), Wt + (size_t)n0 * K + k1, K, 128);
                __builtin_amdgcn_s_wait_tensorcnt((short)2);   // stage i done
            } else {
                __builtin_amdgcn_s_wait_tensorcnt((short)0);
            }
        }
        __syncthreads();

        const unsigned short* Ab = As[cur];
        const unsigned short* Bb = Bs[cur];
        bf16x16 af[2], bf[4];
#pragma unroll
        for (int tr = 0; tr < 2; ++tr) {
            const int row = wr * 32 + tr * 16 + l16;
            Frag u;
            u.q[0] = *(const uint4*)(Ab + row * 40 + lh * 8);
            u.q[1] = *(const uint4*)(Ab + row * 40 + 16 + lh * 8);
            af[tr] = u.v;
        }
#pragma unroll
        for (int tc = 0; tc < 4; ++tc) {
            const int col = wc * 64 + tc * 16 + l16;
            Frag u;
            const uint4* p = (const uint4*)(Bb + col * 40 + lh * 8);
            u.q[0] = p[0];
            u.q[1] = p[2];
            bf[tc] = u.v;
        }
#pragma unroll
        for (int tr = 0; tr < 2; ++tr)
#pragma unroll
            for (int tc = 0; tc < 4; ++tc)
                acc[tr][tc] = WMMA_BF16(af[tr], bf[tc], acc[tr][tc]);
        __syncthreads();   // before buffer `cur` is overwritten by stage i+2 issue
    }

    // ---- epilogue ----
#pragma unroll
    for (int tr = 0; tr < 2; ++tr) {
#pragma unroll
        for (int tc = 0; tc < 4; ++tc) {
            const int col = n0 + wc * 64 + tc * 16 + l16;
            const float bv = bias[col];
#pragma unroll
            for (int i = 0; i < 8; ++i) {
                const int row = m0 + wr * 32 + tr * 16 + i + 8 * lh;
                float v = acc[tr][tc][i] + bv;
                if (MODE == 2) {   // tanh GELU
                    float t = 0.7978845608028654f * (v + 0.044715f * v * v * v);
                    v = 0.5f * v * (1.0f + tanhf(t));
                }
                const size_t idx = (size_t)row * N + col;
                if (MODE == 1) {
                    outF[idx] = v + res[idx];
                } else {
                    outB[idx] = f2b(v);
                }
            }
        }
    }
}

// ---------------------------------------------------------------------------
// Causal flash attention. qkv bf16 [8192][3072] (q|k|v, 16 heads x 64).
// Block = (b, h, 128 queries), 8 waves of 16 queries. KV blocks of 32.
// y bf16 [8192][1024].
// ---------------------------------------------------------------------------
__global__ __launch_bounds__(256) void attn_kernel(const unsigned short* __restrict__ qkv,
                                                   unsigned short* __restrict__ y) {
    __shared__ alignas(16) unsigned short Vt[64 * 32];        // [d][key]
    __shared__ alignas(16) float          Ssc[8][16 * 32];    // per wave: [row][key]
    __shared__ alignas(16) unsigned short Psc[8][16 * 32];    // per wave: [row][key] bf16
    __shared__ alignas(16) float mrow[8][16];
    __shared__ alignas(16) float lrow[8][16];
    __shared__ alignas(16) float arow[8][16];

    const int tid  = threadIdx.x;
    const int lane = tid & 31;
    const int wid  = tid >> 5;
    const int l16  = lane & 15;
    const int lh   = lane >> 4;
    const int qblk = blockIdx.x;
    const int h    = blockIdx.y;
    const int b    = blockIdx.z;
    const int qbase = qblk * 128;
    const int wq    = qbase + wid * 16;                       // this wave's first query
    const int wqs   = __builtin_amdgcn_readfirstlane(wq);     // wave-uniform (scalar branch)
    const size_t base = (size_t)b * T_ * (3 * C_);

    // Q fragments (16 queries x 64 dims = 2 A-fragments), loaded once.
    bf16x16 aq[2];
    {
        const unsigned short* qp = qkv + base + (size_t)(wq + l16) * (3 * C_) + h * D_;
#pragma unroll
        for (int c = 0; c < 2; ++c) {
            Frag u;
            u.q[0] = *(const uint4*)(qp + c * 32 + lh * 8);
            u.q[1] = *(const uint4*)(qp + c * 32 + 16 + lh * 8);
            aq[c] = u.v;
        }
    }
    if (lane < 16) {
        mrow[wid][lane] = -3.0e38f;
        lrow[wid][lane] = 0.0f;
    }
    f32x8 o[4];
#pragma unroll
    for (int j = 0; j < 4; ++j)
#pragma unroll
        for (int i = 0; i < 8; ++i) o[j][i] = 0.0f;

    const int nkb = qbase / 32 + 4;                           // causal KV-block count
    for (int kb = 0; kb < nkb; ++kb) {
        const int k0 = kb * 32;
        __syncthreads();
        // ---- cooperative stage of V^T (32 keys x 64 dims -> Vt[d][key]) ----
        {
            const int key = tid >> 3, dg = tid & 7;
            const unsigned short* vp =
                qkv + base + (size_t)(k0 + key) * (3 * C_) + 2 * C_ + h * D_ + dg * 8;
            uint4 q = *(const uint4*)vp;
            const unsigned short* e = (const unsigned short*)&q;
#pragma unroll
            for (int j = 0; j < 8; ++j) Vt[(dg * 8 + j) * 32 + key] = e[j];
        }
        __syncthreads();

        if (k0 <= wqs + 15) {                                 // scalar causal skip (EXEC stays full)
            // ---- S = Q K^T (scale + mask), via LDS for softmax layout ----
#pragma unroll
            for (int kt = 0; kt < 2; ++kt) {
                f32x8 s;
#pragma unroll
                for (int i = 0; i < 8; ++i) s[i] = 0.0f;
#pragma unroll
                for (int c = 0; c < 2; ++c) {
                    const int key = k0 + kt * 16 + l16;
                    const unsigned short* kp =
                        qkv + base + (size_t)key * (3 * C_) + C_ + h * D_ + c * 32 + lh * 8;
                    Frag u;
                    u.q[0] = ((const uint4*)kp)[0];
                    u.q[1] = ((const uint4*)kp)[2];
                    s = WMMA_BF16(aq[c], u.v, s);
                }
                const int key = k0 + kt * 16 + l16;
#pragma unroll
                for (int i = 0; i < 8; ++i) {
                    const int row = i + 8 * lh;
                    float v = s[i] * 0.125f;                  // 1/sqrt(64)
                    if (key > wq + row) v = -3.0e38f;
                    Ssc[wid][row * 32 + kt * 16 + l16] = v;
                }
            }
            // ---- online softmax, one row per lane (lanes 0..15) ----
            if (lane < 16) {
                const int row = lane;
                float mo = mrow[wid][row], lo = lrow[wid][row];
                float sv[32];
                float mk = -3.0e38f;
#pragma unroll
                for (int j = 0; j < 32; ++j) {
                    sv[j] = Ssc[wid][row * 32 + j];
                    mk = fmaxf(mk, sv[j]);
                }
                const float mn = fmaxf(mo, mk);
                const float al = __expf(mo - mn);
                float sum = 0.0f;
#pragma unroll
                for (int j = 0; j < 32; ++j) {
                    float p = __expf(sv[j] - mn);
                    sum += p;
                    Psc[wid][row * 32 + j] = f2b(p);
                }
                mrow[wid][row] = mn;
                lrow[wid][row] = lo * al + sum;
                arow[wid][row] = al;
            }
            // ---- rescale O accumulators by per-row alpha ----
            {
                float4 a0 = *(const float4*)&arow[wid][8 * lh];
                float4 a1 = *(const float4*)&arow[wid][8 * lh + 4];
                const float sc[8] = {a0.x, a0.y, a0.z, a0.w, a1.x, a1.y, a1.z, a1.w};
#pragma unroll
                for (int j = 0; j < 4; ++j)
#pragma unroll
                    for (int i = 0; i < 8; ++i) o[j][i] *= sc[i];
            }
            // ---- O += P V ----
            {
                Frag up;
                const unsigned short* pp = Psc[wid] + l16 * 32;
                up.q[0] = *(const uint4*)(pp + lh * 8);
                up.q[1] = *(const uint4*)(pp + 16 + lh * 8);
#pragma unroll
                for (int j = 0; j < 4; ++j) {
                    const int d = j * 16 + l16;
                    Frag uv;
                    const uint4* vp = (const uint4*)(Vt + d * 32 + lh * 8);
                    uv.q[0] = vp[0];
                    uv.q[1] = vp[2];
                    o[j] = WMMA_BF16(up.v, uv.v, o[j]);
                }
            }
        }
    }

    // ---- finalize: O /= l, store bf16 ----
    float4 l0 = *(const float4*)&lrow[wid][8 * lh];
    float4 l1 = *(const float4*)&lrow[wid][8 * lh + 4];
    const float inv[8] = {1.0f / l0.x, 1.0f / l0.y, 1.0f / l0.z, 1.0f / l0.w,
                          1.0f / l1.x, 1.0f / l1.y, 1.0f / l1.z, 1.0f / l1.w};
    unsigned short* yb = y + (size_t)b * T_ * C_;
#pragma unroll
    for (int j = 0; j < 4; ++j)
#pragma unroll
        for (int i = 0; i < 8; ++i) {
            const int row = wq + i + 8 * lh;
            const int d   = j * 16 + l16;
            yb[(size_t)row * C_ + h * D_ + d] = f2b(o[j][i] * inv[i]);
        }
}

// ---------------------------------------------------------------------------
// Launch: cvt+transpose weights -> LN1 -> QKV GEMM -> attention ->
//         proj(+res) -> LN2 -> FC1(+GELU) -> FC2(+res) -> d_out
// ---------------------------------------------------------------------------
extern "C" void kernel_launch(void* const* d_in, const int* in_sizes, int n_in,
                              void* d_out, int out_size, void* d_ws, size_t ws_size,
                              hipStream_t stream) {
    const float* x      = (const float*)d_in[0];
    const float* ln1_g  = (const float*)d_in[1];
    const float* ln1_b  = (const float*)d_in[2];
    const float* w_attn = (const float*)d_in[3];
    const float* b_attn = (const float*)d_in[4];
    const float* w_proj = (const float*)d_in[5];
    const float* b_proj = (const float*)d_in[6];
    const float* ln2_g  = (const float*)d_in[7];
    const float* ln2_b  = (const float*)d_in[8];
    const float* w_fc   = (const float*)d_in[9];
    const float* b_fc   = (const float*)d_in[10];
    const float* w_fc2  = (const float*)d_in[11];
    const float* b_fc2  = (const float*)d_in[12];
    float* out = (float*)d_out;

    char* ws = (char*)d_ws;
    size_t off = 0;
    auto alloc = [&](size_t bytes) -> void* {
        void* p = ws + off;
        off += (bytes + 255) & ~(size_t)255;
        return p;
    };
    const size_t nWAttn = (size_t)C_ * 3 * C_;
    const size_t nWProj = (size_t)C_ * C_;
    const size_t nWFc   = (size_t)C_ * 4 * C_;
    const size_t nWFc2  = (size_t)4 * C_ * C_;

    unsigned short* wAttnT = (unsigned short*)alloc(nWAttn * 2);   // [3C][C]
    unsigned short* wProjT = (unsigned short*)alloc(nWProj * 2);   // [C][C]
    unsigned short* wFcT   = (unsigned short*)alloc(nWFc * 2);     // [4C][C]
    unsigned short* wFc2T  = (unsigned short*)alloc(nWFc2 * 2);    // [C][4C]
    unsigned short* actB   = (unsigned short*)alloc((size_t)ROWS * C_ * 2);      // LN1/LN2 out
    unsigned short* yB     = (unsigned short*)alloc((size_t)ROWS * C_ * 2);      // attn out
    unsigned short* qkvB   = (unsigned short*)alloc((size_t)ROWS * 3 * C_ * 2);  // qkv
    unsigned short* hB     = (unsigned short*)alloc((size_t)ROWS * 4 * C_ * 2);  // gelu(fc1)
    float*          x1     = (float*)alloc((size_t)ROWS * C_ * 4);               // x + attn

    // 1) weights -> bf16, transposed to [N][K]
    cvt_t_kernel<<<(int)((nWAttn + 255) / 256), 256, 0, stream>>>(w_attn, wAttnT, C_, 3 * C_);
    cvt_t_kernel<<<(int)((nWProj + 255) / 256), 256, 0, stream>>>(w_proj, wProjT, C_, C_);
    cvt_t_kernel<<<(int)((nWFc + 255) / 256), 256, 0, stream>>>(w_fc, wFcT, C_, 4 * C_);
    cvt_t_kernel<<<(int)((nWFc2 + 255) / 256), 256, 0, stream>>>(w_fc2, wFc2T, 4 * C_, C_);

    // 2) LN1
    ln_kernel<<<ROWS, 256, 0, stream>>>(x, ln1_g, ln1_b, actB);

    // 3) QKV GEMM: [8192,1024] x [1024,3072] -> bf16 qkv
    gemm_bf16<0><<<dim3(3 * C_ / 128, ROWS / 128), 256, 0, stream>>>(
        actB, wAttnT, b_attn, nullptr, nullptr, qkvB, ROWS, 3 * C_, C_);

    // 4) causal flash attention -> bf16 y
    attn_kernel<<<dim3(T_ / 128, H_, B_), 256, 0, stream>>>(qkvB, yB);

    // 5) projection + residual: x1 = x + y @ w_proj + b_proj
    gemm_bf16<1><<<dim3(C_ / 128, ROWS / 128), 256, 0, stream>>>(
        yB, wProjT, b_proj, x, x1, nullptr, ROWS, C_, C_);

    // 6) LN2
    ln_kernel<<<ROWS, 256, 0, stream>>>(x1, ln2_g, ln2_b, actB);

    // 7) FC1 + tanh GELU: [8192,1024] x [1024,4096] -> bf16 h
    gemm_bf16<2><<<dim3(4 * C_ / 128, ROWS / 128), 256, 0, stream>>>(
        actB, wFcT, b_fc, nullptr, nullptr, hB, ROWS, 4 * C_, C_);

    // 8) FC2 + residual: out = x1 + h @ w_fc2 + b_fc2
    gemm_bf16<1><<<dim3(C_ / 128, ROWS / 128), 256, 0, stream>>>(
        hB, wFc2T, b_fc2, x1, out, nullptr, ROWS, C_, 4 * C_);

    (void)in_sizes; (void)n_in; (void)out_size; (void)ws_size;
}
